// MultiHeadAttention_85057532330053
// MI455X (gfx1250) — matrix operand order repaired and checked by reading, not compiled, not run on hardware
//
#include <hip/hip_runtime.h>

typedef unsigned short u16;
typedef __attribute__((ext_vector_type(8)))  float  v8f;
typedef __attribute__((ext_vector_type(16))) __bf16 v16bf;

#define NEG_INF_F (-1.0e9f)

// ---------- bf16 helpers ----------
__device__ __forceinline__ u16 f32_bf16(float f) {
    unsigned u = __float_as_uint(f);
    unsigned r = u + 0x7FFFu + ((u >> 16) & 1u);   // round-to-nearest-even
    return (u16)(r >> 16);
}
__device__ __forceinline__ float bf16_f32(u16 h) {
    return __uint_as_float(((unsigned)h) << 16);
}

// ---------- gfx1250 async Global -> LDS (ASYNCcnt path) ----------
__device__ __forceinline__ unsigned lds_off(const void* p) {
    return (unsigned)(size_t)p;    // generic LDS addr carries raw offset in [31:0]
}
__device__ __forceinline__ void async_b128(unsigned ldsdst, const void* gsrc) {
    asm volatile("global_load_async_to_lds_b128 %0, %1, off"
                 :: "v"(ldsdst), "v"(gsrc) : "memory");
}
__device__ __forceinline__ void wait_async_le(int n) {   // n is 0/2/4 at call sites
    if (n == 4)      asm volatile("s_wait_asynccnt 0x4" ::: "memory");
    else if (n == 2) asm volatile("s_wait_asynccnt 0x2" ::: "memory");
    else             asm volatile("s_wait_asynccnt 0x0" ::: "memory");
}

// ---------- WMMA fragment loaders (wave32, 16x16x32 bf16) ----------
// A fragment: 16x32 (MxK), row-major source, leading dim `ld` (elements).
__device__ __forceinline__ void load_frag_a(const u16* src, int ld, int lane,
                                            uint4& c0, uint4& c1) {
    int row = lane & 15;
    int hi  = (lane >> 4) & 1;
    const u16* p = src + (size_t)row * ld + hi * 8;
    c0 = *(const uint4*)(p);        // K = hi*8 + 0..7
    c1 = *(const uint4*)(p + 16);   // K = 16 + hi*8 + 0..7
}

// B fragment: 32x16 (KxN) with B[k][n] = T[n][k]; T row-major, leading dim `ld`.
__device__ __forceinline__ void load_frag_bT(const u16* src, int ld, int lane,
                                             uint4& c0, uint4& c1) {
    int col = lane & 15;
    int hi  = (lane >> 4) & 1;
    const u16* p = src + (size_t)col * ld + hi * 8;
    c0 = *(const uint4*)(p);        // K = hi*8 + 0..7
    c1 = *(const uint4*)(p + 16);   // K = 16 + hi*8 + 0..7
}

__device__ __forceinline__ v8f wmma_bf16(uint4 a0, uint4 a1, uint4 b0, uint4 b1, v8f c) {
    union Pack { uint4 u[2]; v16bf v; } A, B;
    A.u[0] = a0; A.u[1] = a1;
    B.u[0] = b0; B.u[1] = b1;
    return __builtin_amdgcn_wmma_f32_16x16x32_bf16(false, A.v, false, B.v,
                                                   (short)0, c, false, false);
}

// ---------- elementwise kernels ----------
__global__ void cvt_bf16_kernel(const float* __restrict__ in, u16* __restrict__ out, int n) {
    for (int i = blockIdx.x * blockDim.x + threadIdx.x; i < n; i += gridDim.x * blockDim.x)
        out[i] = f32_bf16(in[i]);
}

// WT[n*K + k] = bf16(W[k*N + n])
__global__ void transpose_cvt_kernel(const float* __restrict__ W, u16* __restrict__ WT,
                                     int K, int N) {
    int total = K * N;
    for (int i = blockIdx.x * blockDim.x + threadIdx.x; i < total; i += gridDim.x * blockDim.x) {
        int k = i % K;
        int n = i / K;
        WT[(size_t)n * K + k] = f32_bf16(W[(size_t)k * N + n]);
    }
}

// RoPE + head-major reshape: in [b, l, H*DH] bf16 -> out [b, H, l, DH] bf16
__global__ void rope_kernel(const u16* __restrict__ in, u16* __restrict__ out,
                            int B, int H, int L, int DH) {
    int total = B * H * L * (DH / 2);
    for (int i = blockIdx.x * blockDim.x + threadIdx.x; i < total; i += gridDim.x * blockDim.x) {
        int t  = i;
        int p2 = t % (DH / 2); t /= (DH / 2);
        int pos = t % L;       t /= L;
        int h   = t % H;
        int b   = t / H;
        float ang = (float)pos * __expf(-((float)(2 * p2) / (float)DH) * 9.21034037197618f);
        float sn, cs;
        __sincosf(ang, &sn, &cs);
        size_t ib = (((size_t)b * L + pos) * H + h) * DH + 2 * p2;
        float x0 = bf16_f32(in[ib]);
        float x1 = bf16_f32(in[ib + 1]);
        size_t ob = (((size_t)b * H + h) * L + pos) * DH + 2 * p2;
        out[ob]     = f32_bf16(x0 * cs - x1 * sn);
        out[ob + 1] = f32_bf16(x1 * cs + x0 * sn);
    }
}

// V transpose: in [b, l, H*DH] -> out [b, H, DH, l]
__global__ void vtrans_kernel(const u16* __restrict__ in, u16* __restrict__ out,
                              int B, int H, int L, int DH) {
    int total = B * H * DH * L;
    for (int i = blockIdx.x * blockDim.x + threadIdx.x; i < total; i += gridDim.x * blockDim.x) {
        int t  = i;
        int pos = t % L;  t /= L;
        int dh  = t % DH; t /= DH;
        int h   = t % H;
        int b   = t / H;
        out[i] = in[(((size_t)b * L + pos) * H + h) * DH + dh];
    }
}

// ---------- bf16 GEMM: C[M,N] = alpha * A[M,K] @ BT[N,K]^T ----------
// Block tile 128x128, BK=32, TRIPLE-buffered LDS fed by async global->LDS
// (one barrier per K-step). 8 waves, each 32x64 (8 v_wmma per K-step).
template <bool OUT_F32>
__global__ __launch_bounds__(256) void gemm_bf16_bt(const u16* __restrict__ A,
                                                    const u16* __restrict__ BT,
                                                    u16* __restrict__ Cbf,
                                                    float* __restrict__ Cf,
                                                    int M, int N, int K, float alpha) {
    __shared__ __align__(16) u16 sA[3][128 * 32];
    __shared__ __align__(16) u16 sB[3][128 * 32];

    const int tid  = threadIdx.x;
    const int lane = tid & 31;
    const int wid  = tid >> 5;
    const int wm   = wid & 3;        // 0..3 -> 32-row strip
    const int wn   = wid >> 2;       // 0..1 -> 64-col strip
    const int nb   = N >> 7;
    const int bm   = blockIdx.x / nb;
    const int bn   = blockIdx.x % nb;

    const unsigned aL[3] = { lds_off(&sA[0][0]), lds_off(&sA[1][0]), lds_off(&sA[2][0]) };
    const unsigned bL[3] = { lds_off(&sB[0][0]), lds_off(&sB[1][0]), lds_off(&sB[2][0]) };

    const int NK = K >> 5;
    auto issue = [&](int kt, int buf) {
        const int kb = kt << 5;
#pragma unroll
        for (int s = 0; s < 2; ++s) {
            int e = (tid + s * 256) * 8;            // element index in 128x32 tile
            int row = e >> 5, col = e & 31;
            async_b128(aL[buf] + e * 2, A  + (size_t)(bm * 128 + row) * K + kb + col);
        }
#pragma unroll
        for (int s = 0; s < 2; ++s) {
            int e = (tid + s * 256) * 8;
            int row = e >> 5, col = e & 31;
            async_b128(bL[buf] + e * 2, BT + (size_t)(bn * 128 + row) * K + kb + col);
        }
    };

    v8f acc[2][4];
    const v8f z = {0.f, 0.f, 0.f, 0.f, 0.f, 0.f, 0.f, 0.f};
#pragma unroll
    for (int i = 0; i < 2; ++i)
#pragma unroll
        for (int t = 0; t < 4; ++t) acc[i][t] = z;

    issue(0, 0);
    if (NK > 1) issue(1, 1);

    for (int kt = 0; kt < NK; ++kt) {
        const int buf = kt % 3;
        wait_async_le((kt + 1 < NK) ? 4 : 0);   // buffers <= kt complete (in-order)
        __syncthreads();                        // tile kt visible; prev compute finished
        if (kt + 2 < NK) issue(kt + 2, (kt + 2) % 3);   // overwrites buffer read at kt-1

        uint4 fa[2][2];
#pragma unroll
        for (int i = 0; i < 2; ++i)
            load_frag_a(&sA[buf][(wm * 32 + i * 16) * 32], 32, lane, fa[i][0], fa[i][1]);
#pragma unroll
        for (int t = 0; t < 4; ++t) {
            uint4 b0, b1;
            load_frag_bT(&sB[buf][(wn * 64 + t * 16) * 32], 32, lane, b0, b1);
#pragma unroll
            for (int i = 0; i < 2; ++i)
                acc[i][t] = wmma_bf16(fa[i][0], fa[i][1], b0, b1, acc[i][t]);
        }
    }

    const int hi = lane >> 4, cl = lane & 15;
#pragma unroll
    for (int i = 0; i < 2; ++i)
#pragma unroll
        for (int t = 0; t < 4; ++t)
#pragma unroll
            for (int r = 0; r < 8; ++r) {
                int row = bm * 128 + wm * 32 + i * 16 + r + 8 * hi;
                int col = bn * 128 + wn * 64 + t * 16 + cl;
                float v = acc[i][t][r] * alpha;
                size_t idx = (size_t)row * N + col;
                if (OUT_F32) Cf[idx] = v;
                else         Cbf[idx] = f32_bf16(v);
            }
}

// ---------- flash attention (causal, online softmax) ----------
// Q,K: [b,H,L,DH] bf16 (RoPE applied). Vt: [b,H,DH,L] bf16. Y: [b,L,H*DH] bf16.
// Block = 8 waves x 16 query rows. Triple-buffered async K/V staging shared by
// all waves; masking only applied on diagonal-crossing tiles.
__global__ __launch_bounds__(256) void attn_kernel(const u16* __restrict__ Q,
                                                   const u16* __restrict__ Kh,
                                                   const u16* __restrict__ Vt,
                                                   u16* __restrict__ Y,
                                                   int B, int H, int L, int DH) {
    __shared__ __align__(16) u16 sK[3][32 * 64];    // [key][dh]
    __shared__ __align__(16) u16 sV[3][64 * 32];    // [dh][key]
    __shared__ __align__(16) u16 pbuf[8][16 * 32];  // per-wave P staging

    const int tid  = threadIdx.x;
    const int lane = tid & 31;
    const int wave = tid >> 5;
    const int qtiles = L / 128;
    const int bh = blockIdx.x / qtiles;
    const int qt = blockIdx.x % qtiles;
    const int b  = bh / H;
    const int h  = bh % H;
    const int qbase = qt * 128 + wave * 16;

    const u16* Qp = Q  + (size_t)bh * L * DH;
    const u16* Kp = Kh + (size_t)bh * L * DH;
    const u16* Vp = Vt + (size_t)bh * DH * L;

    const unsigned kL[3] = { lds_off(&sK[0][0]), lds_off(&sK[1][0]), lds_off(&sK[2][0]) };
    const unsigned vL[3] = { lds_off(&sV[0][0]), lds_off(&sV[1][0]), lds_off(&sV[2][0]) };

    const int NT    = qt * 4 + 4;                 // block-wide key tiles (32 keys each)
    const int nkt_w = (qbase + 16 + 31) >> 5;     // this wave's causal tile bound

    auto issue = [&](int kt, int buf) {
        const int kb = kt << 5;
        {   // K tile: 32 rows x 64 dh
            int e = tid * 8;
            int row = e >> 6, col = e & 63;
            async_b128(kL[buf] + e * 2, Kp + (size_t)(kb + row) * DH + col);
        }
        {   // V tile: 64 dh-rows x 32 keys (from transposed V)
            int e = tid * 8;
            int row = e >> 5, col = e & 31;
            async_b128(vL[buf] + e * 2, Vp + (size_t)row * L + kb + col);
        }
    };

    // Q fragments for both 32-wide K-steps of the 64-deep dot product
    uint4 qa[2][2];
    load_frag_a(Qp + (size_t)qbase * DH + 0,  DH, lane, qa[0][0], qa[0][1]);
    load_frag_a(Qp + (size_t)qbase * DH + 32, DH, lane, qa[1][0], qa[1][1]);

    const v8f z = {0.f, 0.f, 0.f, 0.f, 0.f, 0.f, 0.f, 0.f};
    v8f acc[4];
#pragma unroll
    for (int t = 0; t < 4; ++t) acc[t] = z;
    float m[8], lsum[8];
#pragma unroll
    for (int r = 0; r < 8; ++r) { m[r] = -1.0e30f; lsum[r] = 0.f; }

    const int hi = lane >> 4, cl = lane & 15;

    issue(0, 0);
    if (NT > 1) issue(1, 1);

    for (int kt = 0; kt < NT; ++kt) {
        const int buf = kt % 3;
        const int kb  = kt << 5;
        wait_async_le((kt + 1 < NT) ? 2 : 0);
        __syncthreads();
        if (kt + 2 < NT) issue(kt + 2, (kt + 2) % 3);

        if (kt < nkt_w) {   // this wave still has unmasked keys in the tile
            // S = Q K^T (16 x 32 keys)
            v8f s[2] = {z, z};
#pragma unroll
            for (int t = 0; t < 2; ++t) {
                uint4 b00, b01, b10, b11;
                load_frag_bT(&sK[buf][(t * 16) * 64 + 0],  64, lane, b00, b01);
                load_frag_bT(&sK[buf][(t * 16) * 64 + 32], 64, lane, b10, b11);
                s[t] = wmma_bf16(qa[0][0], qa[0][1], b00, b01, s[t]);
                s[t] = wmma_bf16(qa[1][0], qa[1][1], b10, b11, s[t]);
            }

            // causal mask: only diagonal-crossing tiles need it (scalar branch)
            if (kb + 31 > qbase) {
#pragma unroll
                for (int t = 0; t < 2; ++t)
#pragma unroll
                    for (int r = 0; r < 8; ++r) {
                        int rowg = qbase + r + 8 * hi;
                        int colg = kb + t * 16 + cl;
                        if (colg > rowg) s[t][r] += NEG_INF_F;
                    }
            }

            // online softmax (row = 16-lane group in C/D layout)
            float p0[8], p1[8], corr[8];
#pragma unroll
            for (int r = 0; r < 8; ++r) {
                float v = fmaxf(s[0][r], s[1][r]);
#pragma unroll
                for (int off = 1; off < 16; off <<= 1) v = fmaxf(v, __shfl_xor(v, off, 32));
                float mn = fmaxf(m[r], v);
                corr[r] = __expf(m[r] - mn);
                m[r] = mn;
                p0[r] = __expf(s[0][r] - mn);
                p1[r] = __expf(s[1][r] - mn);
                float rs = p0[r] + p1[r];
#pragma unroll
                for (int off = 1; off < 16; off <<= 1) rs += __shfl_xor(rs, off, 32);
                lsum[r] = lsum[r] * corr[r] + rs;
            }
#pragma unroll
            for (int t = 0; t < 4; ++t)
#pragma unroll
                for (int r = 0; r < 8; ++r) acc[t][r] *= corr[r];

            // P (C/D layout) -> LDS -> A-fragment layout
#pragma unroll
            for (int r = 0; r < 8; ++r) {
                pbuf[wave][(r + 8 * hi) * 32 + cl]      = f32_bf16(p0[r]);
                pbuf[wave][(r + 8 * hi) * 32 + 16 + cl] = f32_bf16(p1[r]);
            }
            asm volatile("s_wait_dscnt 0" ::: "memory");
            uint4 pa0, pa1;
            load_frag_a(&pbuf[wave][0], 32, lane, pa0, pa1);

            // O += P V
#pragma unroll
            for (int t = 0; t < 4; ++t) {
                uint4 vb0, vb1;
                load_frag_bT(&sV[buf][(t * 16) * 32], 32, lane, vb0, vb1);
                acc[t] = wmma_bf16(pa0, pa1, vb0, vb1, acc[t]);
            }
        }
    }

    // epilogue: normalize, write Y [b, l, H*DH]
    float inv[8];
#pragma unroll
    for (int r = 0; r < 8; ++r) inv[r] = 1.0f / lsum[r];
    u16* Yp = Y + (size_t)b * L * H * DH;
#pragma unroll
    for (int t = 0; t < 4; ++t)
#pragma unroll
        for (int r = 0; r < 8; ++r) {
            int rowg = qbase + r + 8 * hi;
            int colg = t * 16 + cl;
            Yp[(size_t)rowg * (H * DH) + h * DH + colg] = f32_bf16(acc[t][r] * inv[r]);
        }
}

// ---------- host launcher ----------
extern "C" void kernel_launch(void* const* d_in, const int* in_sizes, int n_in,
                              void* d_out, int out_size, void* d_ws, size_t ws_size,
                              hipStream_t stream) {
    (void)in_sizes; (void)n_in; (void)out_size; (void)ws_size;
    const float* x  = (const float*)d_in[0];
    // d_in[1] = causal mask (recomputed analytically in-kernel)
    const float* Wq = (const float*)d_in[2];
    const float* Wk = (const float*)d_in[3];
    const float* Wv = (const float*)d_in[4];
    const float* Wo = (const float*)d_in[5];
    float* out = (float*)d_out;

    const int B = 2, L = 2048, D = 1024, H = 16, DH = 64;
    const int M = B * L;                       // 4096

    char* ws = (char*)d_ws;
    const size_t SZ_X = (size_t)M * D * 2;     // 8 MiB bf16 activations
    const size_t SZ_W = (size_t)D * D * 2;     // 2 MiB bf16 weights
    u16* xbf = (u16*)(ws);
    u16* wqT = (u16*)(ws + SZ_X);
    u16* wkT = (u16*)(ws + SZ_X + 1 * SZ_W);
    u16* wvT = (u16*)(ws + SZ_X + 2 * SZ_W);
    u16* woT = (u16*)(ws + SZ_X + 3 * SZ_W);
    u16* qbf = (u16*)(ws + SZ_X + 4 * SZ_W);
    u16* kbf = (u16*)(ws + 2 * SZ_X + 4 * SZ_W);
    u16* vbf = (u16*)(ws + 3 * SZ_X + 4 * SZ_W);
    // buffer reuse (stream-ordered, deterministic):
    u16* qr  = xbf;   // RoPE'd Q [b,H,L,DH]
    u16* kr  = qbf;   // RoPE'd K [b,H,L,DH]
    u16* vtr = kbf;   // V^T     [b,H,DH,L]
    u16* ybf = vbf;   // attention out [b,L,D]

    // 1) precision conversion + weight transposes
    cvt_bf16_kernel<<<4096, 256, 0, stream>>>(x, xbf, M * D);
    transpose_cvt_kernel<<<2048, 256, 0, stream>>>(Wq, wqT, D, D);
    transpose_cvt_kernel<<<2048, 256, 0, stream>>>(Wk, wkT, D, D);
    transpose_cvt_kernel<<<2048, 256, 0, stream>>>(Wv, wvT, D, D);
    transpose_cvt_kernel<<<2048, 256, 0, stream>>>(Wo, woT, D, D);

    // 2) projections (WMMA, 128x128 block tiles). q scaled by 1/sqrt(d) = 1/32.
    const int gemm_blocks = (M / 128) * (D / 128);   // 256
    gemm_bf16_bt<false><<<gemm_blocks, 256, 0, stream>>>(xbf, wqT, qbf, nullptr, M, D, D, 0.03125f);
    gemm_bf16_bt<false><<<gemm_blocks, 256, 0, stream>>>(xbf, wkT, kbf, nullptr, M, D, D, 1.0f);
    gemm_bf16_bt<false><<<gemm_blocks, 256, 0, stream>>>(xbf, wvT, vbf, nullptr, M, D, D, 1.0f);

    // 3) RoPE + layout changes
    rope_kernel<<<2048, 256, 0, stream>>>(qbf, qr, B, H, L, DH);
    rope_kernel<<<2048, 256, 0, stream>>>(kbf, kr, B, H, L, DH);
    vtrans_kernel<<<4096, 256, 0, stream>>>(vbf, vtr, B, H, L, DH);

    // 4) causal flash attention (WMMA + async-LDS staged K/V)
    attn_kernel<<<B * H * (L / 128), 256, 0, stream>>>(qr, kr, vtr, ybf, B, H, L, DH);

    // 5) output projection -> fp32 d_out
    gemm_bf16_bt<true><<<gemm_blocks, 256, 0, stream>>>(ybf, woT, nullptr, out, M, D, D, 1.0f);
}
